// LSTM_Model_34205119545534
// MI455X (gfx1250) — matrix operand order repaired and checked by reading, not compiled, not run on hardware
//
#include <hip/hip_runtime.h>

// CDNA5 / gfx1250: wave32, WMMA 16x16x32 f16->f32
typedef __attribute__((ext_vector_type(16))) _Float16 v16h;
typedef __attribute__((ext_vector_type(8)))  float    v8f;

#define B_ 512
#define S_ 512
#define D_ 64
#define H_ 32
#define G_ 128   // 4*H

static __device__ __forceinline__ float sigmoidf_(float x) {
  return 1.0f / (1.0f + __expf(-x));
}

static __device__ __forceinline__ v8f wmma_f16(v16h a, v16h b, v8f c) {
  // (neg_a, A, neg_b, B, c_mod, C, reuse_a, reuse_b)
  return __builtin_amdgcn_wmma_f32_16x16x32_f16(false, a, false, b, (short)0, c,
                                                false, false);
}

// ---------------------------------------------------------------------------
// Kernel 1: gx0[b,s,:] = x[b,s,:] @ W_ih0^T + (b_ih0 + b_hh0)
// For each s: GEMM M=512 (batch), N=128, K=64. Output stored in raw WMMA
// C-fragment order: gx[s][btile(32)][nt(8)][lane(32)][r(8)]  (f32)
// Block = 128 threads = 4 waves; each wave does a 16-row batch tile.
// ---------------------------------------------------------------------------
__global__ void __launch_bounds__(128)
gx0_kernel(const float* __restrict__ x, const float* __restrict__ Wih0,
           const float* __restrict__ bih0, const float* __restrict__ bhh0,
           float* __restrict__ gx) {
  // W_ih0 pre-swizzled into B-fragment order: [ks][nt][lane(N, half)][i(K)]
  __shared__ __align__(32) _Float16 wfr[2][8][32][16];
  const int tid = threadIdx.x;
  for (int e = tid; e < 2 * 8 * 32 * 16; e += 128) {
    const int ks = e >> 12, nt = (e >> 9) & 7, ln = (e >> 4) & 31, i = e & 15;
    const int kk = ((ln >> 4) << 3) + (i & 7) + ((i >> 3) << 4);
    wfr[ks][nt][ln][i] = (_Float16)Wih0[(nt * 16 + (ln & 15)) * D_ + ks * 32 + kk];
  }
  __syncthreads();

  const int wid  = tid >> 5;
  const int lane = tid & 31;
  const int half = lane >> 4;
  const int m    = lane & 15;
  const int s    = blockIdx.y;
  const int tIdx = blockIdx.x * 4 + wid;   // batch tile 0..31
  const int b    = tIdx * 16 + m;

  // A fragments (16x32 f16) for the two K-steps. Lane layout:
  // lane holds row m=lane%16; elem i -> k = ks*32 + (i%8) + (i/8)*16 + half*8
  v16h a[2];
  const float* xrow = x + ((size_t)b * S_ + s) * D_;
#pragma unroll
  for (int ks = 0; ks < 2; ++ks) {
    const float* p0 = xrow + ks * 32 + half * 8;
    const float* p1 = p0 + 16;
#pragma unroll
    for (int i = 0; i < 8; ++i) {
      a[ks][i]     = (_Float16)p0[i];
      a[ks][8 + i] = (_Float16)p1[i];
    }
  }

  float* outp = gx + (((size_t)s * 32 + tIdx) * 8) * 32 * 8 + lane * 8;
#pragma unroll
  for (int nt = 0; nt < 8; ++nt) {
    const int g = nt * 16 + m;           // gate index (depends only on n=lane%16)
    const float bz = bih0[g] + bhh0[g];
    v8f c;
#pragma unroll
    for (int r = 0; r < 8; ++r) c[r] = bz;
#pragma unroll
    for (int ks = 0; ks < 2; ++ks) {
      v16h bfr = *(const v16h*)&wfr[ks][nt][lane][0];
      c = wmma_f16(a[ks], bfr, c);
    }
    *(v8f*)(outp + nt * 256) = c;        // two global_store_b128
  }
}

// ---------------------------------------------------------------------------
// Kernel 2: fused 2-layer sequential scan + final FC.
// One wave (32 threads) per block, 32 blocks; wave owns batch rows
// [16*blockIdx.x, 16*blockIdx.x+16). Per timestep t:
//   g0 = gx0[t] (C-fragments) + h1 @ Whh0^T  (8 WMMA, K=32)
//   pointwise LSTM update -> h1 (LDS transpose -> A fragment f16)
//   g1 = bias1 + h1 @ Wih1^T + h2 @ Whh1^T   (16 WMMA)
//   pointwise LSTM update -> h2
// After t loop: out[b] = h2_last[b,:] . Wfc + bfc
// ---------------------------------------------------------------------------
__global__ void __launch_bounds__(32)
scan_kernel(const float* __restrict__ gx, const float* __restrict__ Whh0,
            const float* __restrict__ Wih1, const float* __restrict__ Whh1,
            const float* __restrict__ bih1, const float* __restrict__ bhh1,
            const float* __restrict__ Wfc, const float* __restrict__ bfc,
            float* __restrict__ out) {
  __shared__ __align__(32) _Float16 frW0[8][32][16];  // Whh0 B-fragments
  __shared__ __align__(32) _Float16 frW1[8][32][16];  // Wih1 B-fragments
  __shared__ __align__(32) _Float16 frW2[8][32][16];  // Whh1 B-fragments
  __shared__ float bias1[G_];
  __shared__ float h1t[16][32];                       // h1 transpose staging
  __shared__ float h2t[16][32];                       // h2 transpose staging

  const int lane = threadIdx.x & 31;
  for (int e = lane; e < 8 * 32 * 16; e += 32) {
    const int nt = e >> 9, ln = (e >> 4) & 31, i = e & 15;
    const int g = nt * 16 + (ln & 15);
    const int kk = ((ln >> 4) << 3) + (i & 7) + ((i >> 3) << 4);
    frW0[nt][ln][i] = (_Float16)Whh0[g * H_ + kk];
    frW1[nt][ln][i] = (_Float16)Wih1[g * H_ + kk];
    frW2[nt][ln][i] = (_Float16)Whh1[g * H_ + kk];
  }
  for (int g = lane; g < G_; g += 32) bias1[g] = bih1[g] + bhh1[g];
  __syncthreads();

  const int half = lane >> 4;
  const int m    = lane & 15;
  const int tIdx = blockIdx.x;            // batch tile 0..31

  v16h h1f, h2f;
#pragma unroll
  for (int i = 0; i < 16; ++i) { h1f[i] = (_Float16)0.f; h2f[i] = (_Float16)0.f; }
  v8f c1[2], c2[2];
#pragma unroll
  for (int j = 0; j < 2; ++j)
#pragma unroll
    for (int r = 0; r < 8; ++r) { c1[j][r] = 0.f; c2[j][r] = 0.f; }

  for (int t = 0; t < S_; ++t) {
    const float* gp = gx + (size_t)(t * 32 + tIdx) * 2048 + lane * 8;
    if (t + 1 < S_)
      __builtin_prefetch(gx + (size_t)((t + 1) * 32 + tIdx) * 2048 + lane * 8, 0, 1);

    // layer-0 gate accumulators, initialized from precomputed gx fragments
    v8f g0[8];
#pragma unroll
    for (int nt = 0; nt < 8; ++nt) g0[nt] = *(const v8f*)(gp + nt * 256);
#pragma unroll
    for (int nt = 0; nt < 8; ++nt) {
      v16h bfr = *(const v16h*)&frW0[nt][lane][0];
      g0[nt] = wmma_f16(h1f, bfr, g0[nt]);
    }
    // layer-0 pointwise: gates live at gidx = nt*16 + n; hidden h = (nt%2)*16+n
#pragma unroll
    for (int j = 0; j < 2; ++j)
#pragma unroll
      for (int r = 0; r < 8; ++r) {
        const float iv = sigmoidf_(g0[0 + j][r]);
        const float fv = sigmoidf_(g0[2 + j][r]);
        const float gv = tanhf(g0[4 + j][r]);
        const float ov = sigmoidf_(g0[6 + j][r]);
        const float cc = fv * c1[j][r] + iv * gv;
        c1[j][r] = cc;
        h1t[r + 8 * half][j * 16 + m] = ov * tanhf(cc);
      }
    __syncthreads();
    // rebuild h1 as A-fragment (row = lane%16, k = (i%8)+(i/8)*16+half*8)
#pragma unroll
    for (int i = 0; i < 8; ++i) {
      h1f[i]     = (_Float16)h1t[m][half * 8 + i];
      h1f[8 + i] = (_Float16)h1t[m][16 + half * 8 + i];
    }

    // layer-1: g1 = bias1 + h1 @ Wih1^T + h2 @ Whh1^T
    v8f g1[8];
#pragma unroll
    for (int nt = 0; nt < 8; ++nt) {
      const float bz = bias1[nt * 16 + m];
      v8f c;
#pragma unroll
      for (int r = 0; r < 8; ++r) c[r] = bz;
      v16h bf1 = *(const v16h*)&frW1[nt][lane][0];
      c = wmma_f16(h1f, bf1, c);
      v16h bf2 = *(const v16h*)&frW2[nt][lane][0];
      c = wmma_f16(h2f, bf2, c);
      g1[nt] = c;
    }
#pragma unroll
    for (int j = 0; j < 2; ++j)
#pragma unroll
      for (int r = 0; r < 8; ++r) {
        const float iv = sigmoidf_(g1[0 + j][r]);
        const float fv = sigmoidf_(g1[2 + j][r]);
        const float gv = tanhf(g1[4 + j][r]);
        const float ov = sigmoidf_(g1[6 + j][r]);
        const float cc = fv * c2[j][r] + iv * gv;
        c2[j][r] = cc;
        h2t[r + 8 * half][j * 16 + m] = ov * tanhf(cc);
      }
    __syncthreads();
#pragma unroll
    for (int i = 0; i < 8; ++i) {
      h2f[i]     = (_Float16)h2t[m][half * 8 + i];
      h2f[8 + i] = (_Float16)h2t[m][16 + half * 8 + i];
    }
  }

  // Final FC on last h2 (still resident in h2t): out[b] = h2 . Wfc + bfc
  if (lane < 16) {
    float acc = bfc[0];
#pragma unroll
    for (int h = 0; h < H_; ++h) acc += h2t[lane][h] * Wfc[h];
    out[tIdx * 16 + lane] = acc;
  }
}

// ---------------------------------------------------------------------------
extern "C" void kernel_launch(void* const* d_in, const int* in_sizes, int n_in,
                              void* d_out, int out_size, void* d_ws, size_t ws_size,
                              hipStream_t stream) {
  const float* x    = (const float*)d_in[0];
  const float* Wih0 = (const float*)d_in[1];
  const float* Whh0 = (const float*)d_in[2];
  const float* bih0 = (const float*)d_in[3];
  const float* bhh0 = (const float*)d_in[4];
  const float* Wih1 = (const float*)d_in[5];
  const float* Whh1 = (const float*)d_in[6];
  const float* bih1 = (const float*)d_in[7];
  const float* bhh1 = (const float*)d_in[8];
  const float* Wfc  = (const float*)d_in[9];
  const float* bfc  = (const float*)d_in[10];
  float* out = (float*)d_out;

  // gx workspace: 512 * 32 * 8 * 32 * 8 floats = 128 MB (fragment-ordered)
  float* gxw = (float*)d_ws;
  (void)in_sizes; (void)n_in; (void)out_size; (void)ws_size;

  dim3 grid1(8, 512);   // x: btile groups (4 waves each), y: timestep s
  gx0_kernel<<<grid1, 128, 0, stream>>>(x, Wih0, bih0, bhh0, gxw);
  scan_kernel<<<32, 32, 0, stream>>>(gxw, Whh0, Wih1, Whh1, bih1, bhh1,
                                     Wfc, bfc, out);
}
